// ModulatedConv2d_61177514164410
// MI455X (gfx1250) — matrix operand order repaired and checked by reading, not compiled, not run on hardware
//
#include <hip/hip_runtime.h>

typedef __attribute__((ext_vector_type(16))) _Float16 v16h;
typedef __attribute__((ext_vector_type(8)))  _Float16 v8h;
typedef __attribute__((ext_vector_type(8)))  float    v8f;

#define B_    8
#define CIN_  64
#define COUT_ 64
#define SDIM_ 512
#define H_    256
#define W_    256

#define TILE_W 32
#define TILE_H 8
#define LDS_W  34              // TILE_W + halo
#define LDS_H  10              // TILE_H + halo
#define LDS_STRIDE 72          // 64 ch + 8 pad halves -> 144B/cell (16B aligned, bank-spread)

// ---------------------------------------------------------------------------
// Prep: style modulation, demodulation, and f16 weights pre-swizzled into the
// V_WMMA_F32_16X16X32_F16 A-operand order used here (contiguous k, not the ISA table order):
//   lane L: row m = L&15 ; half i: k = i + 16*(L>=16)  (within 32-chunk)
// K ordering: k = tap*64 + ci  (tap = ky*3+kx)
// apack layout: [b][mtile(4)][kstep(18)][lane(32)][i(16)] halves
// ---------------------------------------------------------------------------
__global__ __launch_bounds__(256) void modconv_prep(
    const float* __restrict__ style,      // (B,512)
    const float* __restrict__ weight,     // (1,64,64,3,3)
    const float* __restrict__ mod_weight, // (64,512)
    const float* __restrict__ mod_bias,   // (64)
    _Float16* __restrict__ apack)
{
    __shared__ float s_mod[CIN_];
    __shared__ float s_dem[COUT_];
    const int b   = blockIdx.x;
    const int tid = threadIdx.x;
    const float conv_scale = 0.041666666666666664f;  // 1/sqrt(64*9)
    const float mod_scale  = 0.044194173824159216f;  // 1/sqrt(512)

    if (tid < CIN_) {
        float acc = 0.f;
        const float* st = style + b * SDIM_;
        const float* mw = mod_weight + tid * SDIM_;
        for (int s = 0; s < SDIM_; ++s) acc += st[s] * mw[s];
        s_mod[tid] = acc * mod_scale + mod_bias[tid];
    }
    __syncthreads();
    if (tid < COUT_) {
        float sq = 0.f;
        const float* wp = weight + tid * (CIN_ * 9);
        for (int ci = 0; ci < CIN_; ++ci) {
            float sm = s_mod[ci] * conv_scale;
            for (int t = 0; t < 9; ++t) {
                float w = wp[ci * 9 + t] * sm;
                sq += w * w;
            }
        }
        s_dem[tid] = rsqrtf(sq + 1e-8f);
    }
    __syncthreads();

    const int perB = 4 * 18 * 32 * 16;   // 36864 halves per batch
    for (int f = tid; f < perB; f += 256) {
        int i    = f & 15;
        int lane = (f >> 4) & 31;
        int rest = f >> 9;
        int kc   = rest % 18;
        int mt   = rest / 18;
        int cout = mt * 16 + (lane & 15);
        int kk   = i + ((lane >= 16) ? 16 : 0);
        int k    = kc * 32 + kk;
        int tap  = k >> 6;
        int ci   = k & 63;
        float w = conv_scale * weight[(cout * CIN_ + ci) * 9 + tap] * s_mod[ci] * s_dem[cout];
        apack[b * perB + f] = (_Float16)w;
    }
}

// ---------------------------------------------------------------------------
// Main conv: implicit GEMM. Block = 32x8 pixel tile x 64 couts, 8 waves.
// Wave w: mtile = w&3, pixel half = w>>2 (8 groups of 16 px each).
// B-fragment (contiguous k order used here, same as A): lane L: col n = L&15,
//   half i -> k = i + 16*(L>=16)  => 16 contiguous channels from LDS.
// ---------------------------------------------------------------------------
__global__ __launch_bounds__(256) void modconv_main(
    const float* __restrict__ input,     // (B,64,256,256)
    const _Float16* __restrict__ apack,
    float* __restrict__ out)             // (B,64,256,256)
{
    __shared__ _Float16 sIn[LDS_H * LDS_W * LDS_STRIDE];  // 48960 B

    const int x0  = blockIdx.x * TILE_W;
    const int y0  = blockIdx.y * TILE_H;
    const int b   = blockIdx.z;
    const int tid = threadIdx.x;

    // ---- stage input tile (+1px halo) into LDS, f32 -> f16 ----
    const int total = LDS_H * LDS_W * CIN_;  // 21760
    for (int e = tid; e < total; e += 256) {
        int col = e % LDS_W;
        int t   = e / LDS_W;
        int row = t % LDS_H;
        int ci  = t / LDS_H;
        int gy = y0 + row - 1;
        int gx = x0 + col - 1;
        float v = 0.f;
        if ((unsigned)gy < (unsigned)H_ && (unsigned)gx < (unsigned)W_)
            v = input[((b * CIN_ + ci) * H_ + gy) * W_ + gx];
        sIn[(row * LDS_W + col) * LDS_STRIDE + ci] = (_Float16)v;
    }
    __syncthreads();

    const int wave  = tid >> 5;
    const int lane  = tid & 31;
    const int mtile = wave & 3;
    const int hi16  = (lane >= 16) ? 1 : 0;
    const int n     = lane & 15;

    // A fragments for this wave's 16 couts: resident in VGPRs (18 x v16h)
    v16h areg[18];
    const v16h* ap = (const v16h*)apack + ((b * 4 + mtile) * 18) * 32 + lane;
#pragma unroll
    for (int kc = 0; kc < 18; ++kc) areg[kc] = ap[kc * 32];

#pragma unroll 1
    for (int j = 0; j < 8; ++j) {
        const int nt  = (wave >> 2) * 8 + j;   // 0..15 across the block
        const int px0 = (nt & 1) * 16;
        const int py  = nt >> 1;
        v8f acc = {};
#pragma unroll
        for (int kc = 0; kc < 18; ++kc) {
            const int tap = kc >> 1;
            const int dy  = tap / 3, dx = tap % 3;
            const int ci0 = (kc & 1) * 32 + hi16 * 16;
            const _Float16* p =
                &sIn[((py + dy) * LDS_W + (px0 + n + dx)) * LDS_STRIDE + ci0];
            v8h blo = *(const v8h*)(p);        // 16B aligned ds_load_b128
            v8h bhi = *(const v8h*)(p + 8);
            v16h bm = __builtin_shufflevector(blo, bhi,
                        0,1,2,3,4,5,6,7,8,9,10,11,12,13,14,15);
            acc = __builtin_amdgcn_wmma_f32_16x16x32_f16(
                      false, areg[kc], false, bm, (short)0, acc, false, false);
        }
        // C/D layout (ISA 7.12.2): vgpr r, lane L -> M = r + 8*(L>=16), N = L&15
        const int gx = x0 + px0 + n;
        const int gy = y0 + py;
#pragma unroll
        for (int r = 0; r < 8; ++r) {
            int cout = mtile * 16 + r + hi16 * 8;
            out[((b * COUT_ + cout) * H_ + gy) * W_ + gx] = acc[r];
        }
    }
}

extern "C" void kernel_launch(void* const* d_in, const int* in_sizes, int n_in,
                              void* d_out, int out_size, void* d_ws, size_t ws_size,
                              hipStream_t stream) {
    (void)in_sizes; (void)n_in; (void)out_size; (void)ws_size;
    const float* input      = (const float*)d_in[0];
    const float* style      = (const float*)d_in[1];
    const float* weight     = (const float*)d_in[2];
    const float* mod_weight = (const float*)d_in[3];
    const float* mod_bias   = (const float*)d_in[4];
    float* out      = (float*)d_out;
    _Float16* apack = (_Float16*)d_ws;   // 8*36864 halves = 576 KB

    modconv_prep<<<dim3(B_), 256, 0, stream>>>(style, weight, mod_weight, mod_bias, apack);
    modconv_main<<<dim3(W_ / TILE_W, H_ / TILE_H, B_), 256, 0, stream>>>(input, apack, out);
}
